// MultiheadAttention_43430709297330
// MI455X (gfx1250) — matrix operand order repaired and checked by reading, not compiled, not run on hardware
//
#include <hip/hip_runtime.h>

#define E_DIM 1024
#define NHEAD 16
#define HDIM  64
#define LSEQ  2048
#define BATCH 4
#define MROWS (LSEQ * BATCH)   /* 8192 */
#define BHEAD (BATCH * NHEAD)  /* 64   */

typedef __attribute__((ext_vector_type(16))) _Float16 v16h;
typedef __attribute__((ext_vector_type(8)))  _Float16 v8h;
typedef __attribute__((ext_vector_type(8)))  float    v8f;

#define WAIT_ASYNCCNT(n) asm volatile("s_wait_asynccnt %0" :: "n"(n) : "memory")
#define ASYNC_LOAD_B128(ldsoff, gptr)                                     \
  asm volatile("global_load_async_to_lds_b128 %0, %1, off"                \
               :: "v"(ldsoff), "v"(gptr) : "memory")

static __device__ inline v8f wmma_f16(v16h a, v16h b, v8f c) {
  // D(f32 16x16) = A(f16 16x32) * B(f16 32x16) + C
  return __builtin_amdgcn_wmma_f32_16x16x32_f16(false, a, false, b, (short)0, c,
                                                false, false);
}

// 16-lane (N-group) reductions on the VALU via DPP16 butterflies:
// xor1 = quad_perm[1,0,3,2] (0xB1), xor2 = quad_perm[2,3,0,1] (0x4E),
// xor7 = ROW_HALF_MIRROR (0x141), xor15 = ROW_MIRROR (0x140).
#define DPP_STEP_MAX(x, ctrl)                                                  \
  x = fmaxf(x, __int_as_float(__builtin_amdgcn_update_dpp(                     \
             0, __float_as_int(x), ctrl, 0xF, 0xF, true)))
#define DPP_STEP_ADD(x, ctrl)                                                  \
  x = x + __int_as_float(__builtin_amdgcn_update_dpp(                          \
             0, __float_as_int(x), ctrl, 0xF, 0xF, true))

static __device__ inline float red_max16(float x) {
  DPP_STEP_MAX(x, 0xB1);
  DPP_STEP_MAX(x, 0x4E);
  DPP_STEP_MAX(x, 0x141);
  DPP_STEP_MAX(x, 0x140);
  return x;
}
static __device__ inline float red_sum16(float x) {
  DPP_STEP_ADD(x, 0xB1);
  DPP_STEP_ADD(x, 0x4E);
  DPP_STEP_ADD(x, 0x141);
  DPP_STEP_ADD(x, 0x140);
  return x;
}

// A-fragment (16x32 f16, M=lane%16): halves[0..7]=K[16*hi..+8), halves[8..15]=K[16*hi+8..+8)
static __device__ inline v16h a_frag_f16(const _Float16* rowk, int hi) {
  union { v16h v; v8h h[2]; } u;
  u.h[0] = *(const v8h*)(rowk + 16 * hi);
  u.h[1] = *(const v8h*)(rowk + 16 * hi + 8);
  return u.v;
}

// Same fragment but source row is f32 (convert on the fly)
static __device__ inline v16h a_frag_f32(const float* rowk, int hi) {
  v8f f0 = *(const v8f*)(rowk + 16 * hi);
  v8f f1 = *(const v8f*)(rowk + 16 * hi + 8);
  union { v16h v; v8h h[2]; } u;
  v8h a, b;
#pragma unroll
  for (int i = 0; i < 8; ++i) {
    a[i] = (_Float16)f0[i];
    b[i] = (_Float16)f1[i];
  }
  u.h[0] = a;
  u.h[1] = b;
  return u.v;
}

__global__ void cvt_f32_f16(const float* __restrict__ in, void* __restrict__ outv, int n) {
  _Float16* out = (_Float16*)outv;
  int i = blockIdx.x * blockDim.x + threadIdx.x;
  if (i < n) out[i] = (_Float16)in[i];
}

// ---------------------------------------------------------------------------
// GEMM: Out[m,n] = sum_k X[m,k] * W[n,k] + bias[n]
// Block = 4 waves covering 64 M rows x 64 N cols. B tile (64n x 32k, 4KB)
// staged per block into double-buffered LDS via async copies; each wave owns
// a distinct 16-row M tile (A loaded straight from global, no duplication).
// MODE 0: X=f32 -> QPh[bh][l][d] f16 *0.125   MODE 1: X=f32 -> KPh[bh][l][d]
// MODE 2: X=f32 -> VPT[bh][d][l] f16          MODE 3: X=f16 -> f32 row-major
// ---------------------------------------------------------------------------
template <int MODE>
__global__ __launch_bounds__(128) void gemm_wmma(const void* __restrict__ Xv,
                                                 const void* __restrict__ Wv,
                                                 const float* __restrict__ bias,
                                                 void* __restrict__ Outv) {
  const _Float16* W = (const _Float16*)Wv;
  __shared__ __align__(32) _Float16 Bt[2][64][32];  // [buf][n][k] 4KB each

  const int tid = threadIdx.x;
  const int lane = tid & 31, wv = tid >> 5;
  const int hi = lane >> 4, lo = lane & 15;
  const int mbase = blockIdx.x * 64 + wv * 16;
  const int nbase = blockIdx.y * 64;

  auto issueB = [&](int s) {
    const int buf = s & 1;
    const int kb = s * 32;
#pragma unroll
    for (int i = 0; i < 2; ++i) {
      const int idx = tid + i * 128;  // 0..255 16B chunks
      const int rn = idx >> 2;        // n row 0..63
      const int c = (idx & 3) * 8;    // half offset
      const _Float16* g = W + (size_t)(nbase + rn) * E_DIM + kb + c;
      const unsigned l = (unsigned)(unsigned long long)&Bt[buf][rn][c];
      ASYNC_LOAD_B128(l, g);
    }
  };

  v8f acc[4];
#pragma unroll
  for (int t = 0; t < 4; ++t)
#pragma unroll
    for (int r = 0; r < 8; ++r) acc[t][r] = 0.f;

  const size_t arow = (size_t)(mbase + lo) * E_DIM;

  issueB(0);
  const int NK = E_DIM / 32;
  for (int s = 0; s < NK; ++s) {
    if (s + 1 < NK) {
      issueB(s + 1);
      WAIT_ASYNCCNT(2);
    } else {
      WAIT_ASYNCCNT(0);
    }
    __syncthreads();
    const int kb = s * 32, buf = s & 1;
    v16h a = (MODE == 3) ? a_frag_f16((const _Float16*)Xv + arow + kb, hi)
                         : a_frag_f32((const float*)Xv + arow + kb, hi);
    v16h b0 = *(const v16h*)&Bt[buf][0 + lo][16 * hi];
    v16h b1 = *(const v16h*)&Bt[buf][16 + lo][16 * hi];
    v16h b2 = *(const v16h*)&Bt[buf][32 + lo][16 * hi];
    v16h b3 = *(const v16h*)&Bt[buf][48 + lo][16 * hi];
    acc[0] = wmma_f16(a, b0, acc[0]);
    acc[1] = wmma_f16(a, b1, acc[1]);
    acc[2] = wmma_f16(a, b2, acc[2]);
    acc[3] = wmma_f16(a, b3, acc[3]);
    __syncthreads();
  }

#pragma unroll
  for (int t = 0; t < 4; ++t) {
    const int n = nbase + t * 16 + lo;
    const float bv = bias[n];
#pragma unroll
    for (int r = 0; r < 8; ++r) {
      const float val = acc[t][r] + bv;
      const int m = mbase + r + 8 * hi;
      if (MODE == 3) {
        ((float*)Outv)[(size_t)m * E_DIM + n] = val;
      } else {
        const int l_ = m >> 2, b_ = m & 3;     // m = l*BATCH + b
        const int h_ = n >> 6, d_ = n & 63;    // n = h*HDIM + d
        const int bh = b_ * NHEAD + h_;
        _Float16* O = (_Float16*)Outv;
        if (MODE == 0)
          O[((size_t)bh * LSEQ + l_) * HDIM + d_] = (_Float16)(val * 0.125f);
        else if (MODE == 1)
          O[((size_t)bh * LSEQ + l_) * HDIM + d_] = (_Float16)val;
        else
          O[((size_t)bh * HDIM + d_) * LSEQ + l_] = (_Float16)val;
      }
    }
  }
}

// ---------------------------------------------------------------------------
// Flash attention, 4 waves / block. Each block owns 64 q rows of one head;
// K/V tiles (64 cols) are staged into LDS with double-buffered async copies
// (ASYNCcnt) and shared by all 4 waves. 16 WMMAs per wave per stage.
// ---------------------------------------------------------------------------
__global__ __launch_bounds__(128) void flash_wmma(const void* __restrict__ QPv,
                                                  const void* __restrict__ KPv,
                                                  const void* __restrict__ VPTv,
                                                  void* __restrict__ AttnHv) {
  const _Float16* QP = (const _Float16*)QPv;
  const _Float16* KP = (const _Float16*)KPv;
  const _Float16* VPT = (const _Float16*)VPTv;
  _Float16* AttnH = (_Float16*)AttnHv;

  __shared__ __align__(32) _Float16 Kt[2][64][64];   // [buf][j][d]   8KB each
  __shared__ __align__(32) _Float16 Vt[2][64][64];   // [buf][d][j]   8KB each
  __shared__ __align__(32) _Float16 pb[4][16 * 40];  // per-wave P staging

  const int bh = blockIdx.x;
  const int tid = threadIdx.x;
  const int wv = tid >> 5;
  const int lane = tid & 31;
  const int hi = lane >> 4, lo = lane & 15;
  const int l0 = blockIdx.y * 64 + wv * 16;

  // Q A-fragments (pre-scaled by 1/sqrt(HDIM) at projection time)
  const _Float16* qrow = QP + ((size_t)bh * LSEQ + l0 + lo) * HDIM;
  const v16h qa0 = a_frag_f16(qrow, hi);       // d 0..31
  const v16h qa1 = a_frag_f16(qrow + 32, hi);  // d 32..63

  float mrun[8], lrun[8];
  v8f o[4];
#pragma unroll
  for (int r = 0; r < 8; ++r) { mrun[r] = -1e30f; lrun[r] = 0.f; }
#pragma unroll
  for (int t = 0; t < 4; ++t)
#pragma unroll
    for (int r = 0; r < 8; ++r) o[t][r] = 0.f;

  // Issue one 64-column stage: 8KB K + 8KB V, 8 async b128 per thread.
  auto issue = [&](int s) {
    const int buf = s & 1;
    const int j0 = s * 64;
#pragma unroll
    for (int i = 0; i < 4; ++i) {
      const int idx = tid + i * 128;      // 0..511 -> 16B chunk id
      const int r = idx >> 3;             // row 0..63
      const int c = (idx & 7) * 8;        // half offset within row
      const _Float16* gk = KP + ((size_t)bh * LSEQ + j0 + r) * HDIM + c;
      const unsigned lk = (unsigned)(unsigned long long)&Kt[buf][r][c];
      ASYNC_LOAD_B128(lk, gk);
      const _Float16* gv = VPT + ((size_t)bh * HDIM + r) * LSEQ + j0 + c;
      const unsigned lv = (unsigned)(unsigned long long)&Vt[buf][r][c];
      ASYNC_LOAD_B128(lv, gv);
    }
  };

  issue(0);
  const int NST = LSEQ / 64;
  for (int s = 0; s < NST; ++s) {
    if (s + 1 < NST) {
      issue(s + 1);
      WAIT_ASYNCCNT(8);   // just-issued 8 may be in flight; stage s is complete
    } else {
      WAIT_ASYNCCNT(0);
    }
    __syncthreads();
    const int buf = s & 1;

#pragma unroll
    for (int half = 0; half < 2; ++half) {
      const int jj = half * 32;
      // Hoist all K fragments, then the 4 WMMAs for the two S tiles.
      const _Float16* c0 = &Kt[buf][jj + lo][16 * hi];
      const _Float16* c1 = &Kt[buf][jj + 16 + lo][16 * hi];
      const v16h b00 = *(const v16h*)(c0);
      const v16h b01 = *(const v16h*)(c0 + 32);
      const v16h b10 = *(const v16h*)(c1);
      const v16h b11 = *(const v16h*)(c1 + 32);
      v8f s0, s1;
#pragma unroll
      for (int r = 0; r < 8; ++r) { s0[r] = 0.f; s1[r] = 0.f; }
      s0 = wmma_f16(qa0, b00, s0);
      s1 = wmma_f16(qa0, b10, s1);
      s0 = wmma_f16(qa1, b01, s0);
      s1 = wmma_f16(qa1, b11, s1);

      // Online softmax; per-lane r holds row m = r + 8*hi, N = lane%16.
#pragma unroll
      for (int r = 0; r < 8; ++r) {
        const float sm = red_max16(fmaxf(s0[r], s1[r]));
        const float mn = fmaxf(mrun[r], sm);
        const float sc = __expf(mrun[r] - mn);
        mrun[r] = mn;
        const float p0 = __expf(s0[r] - mn);
        const float p1 = __expf(s1[r] - mn);
        lrun[r] = lrun[r] * sc + red_sum16(p0 + p1);
#pragma unroll
        for (int t = 0; t < 4; ++t) o[t][r] *= sc;
        const int mr = r + 8 * hi;
        pb[wv][mr * 40 + lo] = (_Float16)p0;
        pb[wv][mr * 40 + 16 + lo] = (_Float16)p1;
      }

      // Reshape P (C-fragment -> A-fragment) through per-wave LDS
      const v16h pa = a_frag_f16(&pb[wv][lo * 40], hi);

      // O += P * V  (Vt holds V^T tile: [d][j])
      const v16h vb0 = *(const v16h*)&Vt[buf][0 + lo][jj + 16 * hi];
      const v16h vb1 = *(const v16h*)&Vt[buf][16 + lo][jj + 16 * hi];
      const v16h vb2 = *(const v16h*)&Vt[buf][32 + lo][jj + 16 * hi];
      const v16h vb3 = *(const v16h*)&Vt[buf][48 + lo][jj + 16 * hi];
      o[0] = wmma_f16(pa, vb0, o[0]);
      o[1] = wmma_f16(pa, vb1, o[1]);
      o[2] = wmma_f16(pa, vb2, o[2]);
      o[3] = wmma_f16(pa, vb3, o[3]);
    }
    __syncthreads();  // all waves done with buf before it is refilled
  }

  // Normalize and scatter to (m = l*BATCH + b, e = h*HDIM + d) layout, f16.
  const int b_ = bh >> 4, h_ = bh & 15;
#pragma unroll
  for (int t = 0; t < 4; ++t) {
#pragma unroll
    for (int r = 0; r < 8; ++r) {
      const float val = o[t][r] / lrun[r];
      const int lg = l0 + r + 8 * hi;
      const int col = h_ * HDIM + t * 16 + lo;
      AttnH[((size_t)lg * BATCH + b_) * E_DIM + col] = (_Float16)val;
    }
  }
}

// ---------------------------------------------------------------------------
extern "C" void kernel_launch(void* const* d_in, const int* in_sizes, int n_in,
                              void* d_out, int out_size, void* d_ws, size_t ws_size,
                              hipStream_t stream) {
  (void)in_sizes; (void)n_in; (void)out_size; (void)ws_size;
  const float* q   = (const float*)d_in[0];
  const float* k   = (const float*)d_in[1];
  const float* v   = (const float*)d_in[2];
  const float* ipw = (const float*)d_in[3];  // (3E, E)
  const float* ipb = (const float*)d_in[4];  // (3E,)
  const float* ow  = (const float*)d_in[5];  // (E, E)
  const float* ob  = (const float*)d_in[6];  // (E,)

  char* ws = (char*)d_ws;
  size_t off = 0;
  void* Wh   = ws + off; off += (size_t)3 * E_DIM * E_DIM * 2;      // 6 MB f16
  void* OWh  = ws + off; off += (size_t)E_DIM * E_DIM * 2;          // 2 MB f16
  void* QPh  = ws + off; off += (size_t)BHEAD * LSEQ * HDIM * 2;    // 16 MB
  void* KPh  = ws + off; off += (size_t)BHEAD * LSEQ * HDIM * 2;    // 16 MB
  void* VPT  = ws + off; off += (size_t)BHEAD * LSEQ * HDIM * 2;    // 16 MB
  void* Attn = ws + off; off += (size_t)MROWS * E_DIM * 2;          // 16 MB

  {
    int n = 3 * E_DIM * E_DIM;
    cvt_f32_f16<<<(n + 255) / 256, 256, 0, stream>>>(ipw, Wh, n);
    n = E_DIM * E_DIM;
    cvt_f32_f16<<<(n + 255) / 256, 256, 0, stream>>>(ow, OWh, n);
  }

  dim3 gg(MROWS / 64, E_DIM / 64), gb(128);
  gemm_wmma<0><<<gg, gb, 0, stream>>>(q, Wh, ipb, QPh);
  gemm_wmma<1><<<gg, gb, 0, stream>>>(k, (char*)Wh + (size_t)E_DIM * E_DIM * 2,
                                      ipb + E_DIM, KPh);
  gemm_wmma<2><<<gg, gb, 0, stream>>>(v, (char*)Wh + (size_t)2 * E_DIM * E_DIM * 2,
                                      ipb + 2 * E_DIM, VPT);
  flash_wmma<<<dim3(BHEAD, LSEQ / 64), 128, 0, stream>>>(QPh, KPh, VPT, Attn);
  gemm_wmma<3><<<gg, gb, 0, stream>>>(Attn, OWh, ob, d_out);
}
